// MoEBlock_27230092657640
// MI455X (gfx1250) — compile-verified, matrix operand order checked by reading, not run on hardware
//
#include <hip/hip_runtime.h>
#include <hip/hip_bf16.h>
#include <math.h>

#define N_TOK 2048
#define D_DIM 1024
#define F_DIM 2048
#define E_NUM 8
#define K_SEL 2

typedef __attribute__((ext_vector_type(16))) __bf16 v16bf;
typedef __attribute__((ext_vector_type(8)))  __bf16 v8bf;
typedef __attribute__((ext_vector_type(4)))  __bf16 v4bf;
typedef __attribute__((ext_vector_type(8)))  float  v8f;
typedef __attribute__((ext_vector_type(4)))  int    v4i;

// ---------------- async global->LDS (CDNA5) with fallback ----------------
#if defined(__has_builtin)
#  if __has_builtin(__builtin_amdgcn_global_load_async_to_lds_b128) && \
      __has_builtin(__builtin_amdgcn_s_wait_asynccnt)
#    define USE_ASYNC 1
#  endif
#endif

typedef __attribute__((address_space(1))) v4i gv4i;
typedef __attribute__((address_space(3))) v4i lv4i;

__device__ __forceinline__ void cp16(void* lds_dst, const void* gsrc) {
#if USE_ASYNC
    __builtin_amdgcn_global_load_async_to_lds_b128(
        (gv4i*)(uintptr_t)gsrc,
        (lv4i*)(uint32_t)(uintptr_t)lds_dst,
        0, 0);
#else
    *(v8bf*)lds_dst = *(const v8bf*)gsrc;
#endif
}

// ---------------- workspace layout (all offsets 256B-aligned) ----------------
constexpr size_t OFF_COUNTS  = 0;                                         // E ints
constexpr size_t OFF_OFFSETS = 256;                                       // E ints
constexpr size_t OFF_TOK     = 512;                                       // E*N ints
constexpr size_t OFF_SLOT    = OFF_TOK  + (size_t)E_NUM * N_TOK * 4;      // E*N ints
constexpr size_t OFF_WGT     = OFF_SLOT + (size_t)E_NUM * N_TOK * 4;      // E*N f32
constexpr size_t OFF_XBF     = OFF_WGT  + (size_t)E_NUM * N_TOK * 4;      // N*D bf16
constexpr size_t OFF_INNER   = OFF_XBF  + (size_t)N_TOK * D_DIM * 2;      // (K*N)*F bf16
constexpr size_t OFF_SLOTOUT = OFF_INNER + (size_t)K_SEL * N_TOK * F_DIM * 2; // K*N*D f32
constexpr size_t OFF_WIT     = OFF_SLOTOUT + (size_t)K_SEL * N_TOK * D_DIM * 4; // [E][F][D] bf16
constexpr size_t OFF_WST     = OFF_WIT + (size_t)E_NUM * D_DIM * F_DIM * 2;     // [E][F][D] bf16
constexpr size_t OFF_WOT     = OFF_WST + (size_t)E_NUM * D_DIM * F_DIM * 2;     // [E][D][F] bf16

// ---------------- fragment load helper ----------------
// Tile rows: 32 contiguous bf16 along K. Per ISA 7.12.2 (16-bit A/B, 16x16x32):
// lanes 0-15 hold K 0..7 / 16..23; lanes 16-31 hold K 8..15 / 24..31.
__device__ __forceinline__ v16bf ld_frag(const __bf16* p, int lane) {
    const int k0 = (lane < 16) ? 0 : 8;
    v8bf lo = *(const v8bf*)(p + k0);
    v8bf hi = *(const v8bf*)(p + k0 + 16);
    v16bf r;
    #pragma unroll
    for (int i = 0; i < 8; ++i) { r[i] = lo[i]; r[8 + i] = hi[i]; }
    return r;
}

__device__ __forceinline__ float gelu_exact(float h) {
    return 0.5f * h * (1.0f + erff(h * 0.70710678118654752f));
}

// ---------------- kernels ----------------
__global__ void k_init(int* counts) {
    if (threadIdx.x < E_NUM) counts[threadIdx.x] = 0;
}

__global__ __launch_bounds__(256) void k_convert(const float* __restrict__ x,
                                                 __bf16* __restrict__ xb) {
    size_t i = ((size_t)blockIdx.x * blockDim.x + threadIdx.x) * 4;
    float4 v = *(const float4*)(x + i);
    v4bf o;
    o[0] = (__bf16)v.x; o[1] = (__bf16)v.y; o[2] = (__bf16)v.z; o[3] = (__bf16)v.w;
    *(v4bf*)(xb + i) = o;
}

// transpose + convert one [R][C] f32 matrix (per expert via blockIdx.z) to [C][R] bf16
__global__ __launch_bounds__(256) void k_transpose_cvt(const float* __restrict__ in,
                                                       __bf16* __restrict__ out,
                                                       int R, int C) {
    __shared__ float tile[32][33];
    const int r0 = blockIdx.y * 32, c0 = blockIdx.x * 32;
    const float* src = in + (size_t)blockIdx.z * R * C;
    __bf16* dst = out + (size_t)blockIdx.z * R * C;
    const int tx = threadIdx.x & 31, ty = threadIdx.x >> 5;
    #pragma unroll
    for (int yy = ty; yy < 32; yy += 8)
        tile[yy][tx] = src[(size_t)(r0 + yy) * C + c0 + tx];
    __syncthreads();
    #pragma unroll
    for (int yy = ty; yy < 32; yy += 8)
        dst[(size_t)(c0 + yy) * R + r0 + tx] = (__bf16)tile[tx][yy];
}

// one wave32 per token: logits, softmax, top-2, renorm, append to expert lists
__global__ __launch_bounds__(256) void k_router(const float* __restrict__ states,
                                                const float* __restrict__ Wg,
                                                int* __restrict__ counts,
                                                int* __restrict__ tok_list,
                                                int* __restrict__ slot_list,
                                                float* __restrict__ w_list) {
    const int lane = threadIdx.x & 31;
    const int wv   = threadIdx.x >> 5;
    const int n    = blockIdx.x * 8 + wv;

    float acc[E_NUM];
    #pragma unroll
    for (int e = 0; e < E_NUM; ++e) acc[e] = 0.f;

    const float* x = states + (size_t)n * D_DIM;
    for (int d = lane; d < D_DIM; d += 32) {
        float xv = x[d];
        #pragma unroll
        for (int e = 0; e < E_NUM; ++e) acc[e] += xv * Wg[d * E_NUM + e];
    }
    #pragma unroll
    for (int e = 0; e < E_NUM; ++e) {
        #pragma unroll
        for (int off = 16; off > 0; off >>= 1)
            acc[e] += __shfl_down(acc[e], off, 32);
    }
    if (lane == 0) {
        float m = acc[0];
        #pragma unroll
        for (int e = 1; e < E_NUM; ++e) m = fmaxf(m, acc[e]);
        float p[E_NUM];
        #pragma unroll
        for (int e = 0; e < E_NUM; ++e) p[e] = __expf(acc[e] - m);
        int i0 = 0;
        #pragma unroll
        for (int e = 1; e < E_NUM; ++e) if (p[e] > p[i0]) i0 = e;
        int i1 = (i0 == 0) ? 1 : 0;
        #pragma unroll
        for (int e = 0; e < E_NUM; ++e) if (e != i0 && p[e] > p[i1]) i1 = e;
        float ws = p[i0] + p[i1];
        float w0 = p[i0] / ws, w1 = p[i1] / ws;

        int pos0 = atomicAdd(&counts[i0], 1);
        tok_list[i0 * N_TOK + pos0]  = n;
        slot_list[i0 * N_TOK + pos0] = 0;
        w_list[i0 * N_TOK + pos0]    = w0;

        int pos1 = atomicAdd(&counts[i1], 1);
        tok_list[i1 * N_TOK + pos1]  = n;
        slot_list[i1 * N_TOK + pos1] = 1;
        w_list[i1 * N_TOK + pos1]    = w1;
    }
}

__global__ void k_scan(const int* __restrict__ counts, int* __restrict__ offsets) {
    if (threadIdx.x == 0 && blockIdx.x == 0) {
        int s = 0;
        for (int e = 0; e < E_NUM; ++e) { offsets[e] = s; s += counts[e]; }
    }
}

// pass A: inner = gelu(X_e @ W_in + b_in) * (X_e @ W_scale + b_scale)
// weights pre-transposed+converted: WiT/WsT are [E][F][D] bf16 (K=D contiguous)
// Double-buffered LDS, async loads overlapped with WMMA of previous chunk.
__global__ __launch_bounds__(256) void k_passA(const __bf16* __restrict__ WiT,
                                               const float* __restrict__ b_in,
                                               const __bf16* __restrict__ WsT,
                                               const float* __restrict__ b_scale,
                                               const __bf16* __restrict__ x_bf,
                                               const int* __restrict__ counts,
                                               const int* __restrict__ offsets,
                                               const int* __restrict__ tok_list,
                                               __bf16* __restrict__ inner) {
    const int e   = blockIdx.z;
    const int cnt = counts[e];
    const int m0  = blockIdx.y * 128;
    if (m0 >= cnt) return;
    const int f_base = blockIdx.x * 128;

    __shared__ __align__(16) __bf16 sX [2][128 * 32];
    __shared__ __align__(16) __bf16 sWi[2][128 * 32];
    __shared__ __align__(16) __bf16 sWs[2][128 * 32];
    __shared__ int sTok[128];

    const int tid  = threadIdx.x;
    const int lane = tid & 31;
    const int wv   = tid >> 5;

    if (tid < 128) {
        int rr = m0 + tid;
        sTok[tid] = tok_list[e * N_TOK + ((rr < cnt) ? rr : (cnt - 1))];
    }
    __syncthreads();

    v8f accH[8], accS[8];
    v8f z = {0.f, 0.f, 0.f, 0.f, 0.f, 0.f, 0.f, 0.f};
    #pragma unroll
    for (int c = 0; c < 8; ++c) { accH[c] = z; accS[c] = z; }

    const __bf16* Wi = WiT + (size_t)e * D_DIM * F_DIM;
    const __bf16* Ws = WsT + (size_t)e * D_DIM * F_DIM;

    const int r = tid >> 1;              // staging row 0..127
    const int h = (tid & 1) * 16;        // 16-bf16 half within 32-wide row

    // issue 6 async b128 loads for one 32-wide K chunk into buffer `buf`
    auto stage = [&](int buf, int d0) {
        const __bf16* gx = x_bf + (size_t)sTok[r] * D_DIM + d0 + h;
        cp16(&sX[buf][r * 32 + h],     gx);
        cp16(&sX[buf][r * 32 + h + 8], gx + 8);
        const __bf16* gi = Wi + (size_t)(f_base + r) * D_DIM + d0 + h;
        cp16(&sWi[buf][r * 32 + h],     gi);
        cp16(&sWi[buf][r * 32 + h + 8], gi + 8);
        const __bf16* gs = Ws + (size_t)(f_base + r) * D_DIM + d0 + h;
        cp16(&sWs[buf][r * 32 + h],     gs);
        cp16(&sWs[buf][r * 32 + h + 8], gs + 8);
    };

    stage(0, 0);                          // prologue: chunk 0 in flight

    for (int d0 = 0; d0 < D_DIM; d0 += 32) {
        const int cur = (d0 >> 5) & 1;
        __syncthreads();                  // all waves done reading buf cur^1
        if (d0 + 32 < D_DIM) {
            stage(cur ^ 1, d0 + 32);      // prefetch next chunk (overlaps compute)
#if USE_ASYNC
            __builtin_amdgcn_s_wait_asynccnt(6);   // only chunk-cur loads must drain
#endif
        } else {
#if USE_ASYNC
            __builtin_amdgcn_s_wait_asynccnt(0);
#endif
        }
        __syncthreads();                  // chunk-cur resident for all waves

        const __bf16* bX = sX[cur];
        const __bf16* bI = sWi[cur];
        const __bf16* bS = sWs[cur];
        v16bf a  = ld_frag(bX + (wv * 16 + (lane & 15)) * 32, lane);
        v16bf bi = ld_frag(bI + ((lane & 15)) * 32, lane);
        v16bf bs = ld_frag(bS + ((lane & 15)) * 32, lane);
        #pragma unroll
        for (int c = 0; c < 8; ++c) {
            v16bf bi_n = bi, bs_n = bs;
            if (c < 7) {                  // software-pipeline next col-tile frags
                bi_n = ld_frag(bI + ((c + 1) * 16 + (lane & 15)) * 32, lane);
                bs_n = ld_frag(bS + ((c + 1) * 16 + (lane & 15)) * 32, lane);
            }
            accH[c] = __builtin_amdgcn_wmma_f32_16x16x32_bf16(false, a, false, bi,
                                                              (short)0, accH[c], false, false);
            accS[c] = __builtin_amdgcn_wmma_f32_16x16x32_bf16(false, a, false, bs,
                                                              (short)0, accS[c], false, false);
            bi = bi_n; bs = bs_n;
        }
    }

    const int row_off = offsets[e];
    const int nIdx = lane & 15;
    const int mOff = (lane < 16) ? 0 : 8;
    #pragma unroll
    for (int c = 0; c < 8; ++c) {
        int f = f_base + c * 16 + nIdx;
        float bi = b_in[e * F_DIM + f];
        float bs = b_scale[e * F_DIM + f];
        #pragma unroll
        for (int i = 0; i < 8; ++i) {
            int row = m0 + wv * 16 + mOff + i;
            if (row < cnt) {
                float hh = accH[c][i] + bi;
                float ss = accS[c][i] + bs;
                inner[(size_t)(row_off + row) * F_DIM + f] = (__bf16)(gelu_exact(hh) * ss);
            }
        }
    }
}

// pass B: y = inner @ W_out + b_out, scaled by routing weight, scatter to slot buf
// WoT is [E][D][F] bf16 (K=F contiguous). Double-buffered like pass A.
__global__ __launch_bounds__(256) void k_passB(const __bf16* __restrict__ WoT,
                                               const float* __restrict__ b_out,
                                               const __bf16* __restrict__ inner,
                                               const int* __restrict__ counts,
                                               const int* __restrict__ offsets,
                                               const int* __restrict__ tok_list,
                                               const int* __restrict__ slot_list,
                                               const float* __restrict__ w_list,
                                               float* __restrict__ slot_out) {
    const int e   = blockIdx.z;
    const int cnt = counts[e];
    const int m0  = blockIdx.y * 128;
    if (m0 >= cnt) return;
    const int d_base = blockIdx.x * 128;

    __shared__ __align__(16) __bf16 sA [2][128 * 32];
    __shared__ __align__(16) __bf16 sWo[2][128 * 32];
    __shared__ int   sTok[128];
    __shared__ int   sSlot[128];
    __shared__ float sWt[128];

    const int tid  = threadIdx.x;
    const int lane = tid & 31;
    const int wv   = tid >> 5;

    if (tid < 128) {
        int rr = m0 + tid;
        int ok = (rr < cnt);
        int idx = e * N_TOK + (ok ? rr : 0);
        sTok[tid]  = tok_list[idx];
        sSlot[tid] = slot_list[idx];
        sWt[tid]   = ok ? w_list[idx] : 0.f;
    }

    const int row_off = offsets[e];
    const __bf16* Wo = WoT + (size_t)e * F_DIM * D_DIM;

    v8f acc[8];
    v8f z = {0.f, 0.f, 0.f, 0.f, 0.f, 0.f, 0.f, 0.f};
    #pragma unroll
    for (int c = 0; c < 8; ++c) acc[c] = z;

    const int r = tid >> 1;
    const int h = (tid & 1) * 16;
    int ra = m0 + r; if (ra >= cnt) ra = cnt - 1;   // clamp tail rows of inner

    auto stage = [&](int buf, int f0) {
        const __bf16* ga = inner + (size_t)(row_off + ra) * F_DIM + f0 + h;
        cp16(&sA[buf][r * 32 + h],     ga);
        cp16(&sA[buf][r * 32 + h + 8], ga + 8);
        const __bf16* gw = Wo + (size_t)(d_base + r) * F_DIM + f0 + h;
        cp16(&sWo[buf][r * 32 + h],     gw);
        cp16(&sWo[buf][r * 32 + h + 8], gw + 8);
    };

    stage(0, 0);

    for (int f0 = 0; f0 < F_DIM; f0 += 32) {
        const int cur = (f0 >> 5) & 1;
        __syncthreads();
        if (f0 + 32 < F_DIM) {
            stage(cur ^ 1, f0 + 32);
#if USE_ASYNC
            __builtin_amdgcn_s_wait_asynccnt(4);
#endif
        } else {
#if USE_ASYNC
            __builtin_amdgcn_s_wait_asynccnt(0);
#endif
        }
        __syncthreads();

        const __bf16* bA = sA[cur];
        const __bf16* bW = sWo[cur];
        v16bf a = ld_frag(bA + (wv * 16 + (lane & 15)) * 32, lane);
        v16bf b = ld_frag(bW + ((lane & 15)) * 32, lane);
        #pragma unroll
        for (int c = 0; c < 8; ++c) {
            v16bf b_n = b;
            if (c < 7)
                b_n = ld_frag(bW + ((c + 1) * 16 + (lane & 15)) * 32, lane);
            acc[c] = __builtin_amdgcn_wmma_f32_16x16x32_bf16(false, a, false, b,
                                                             (short)0, acc[c], false, false);
            b = b_n;
        }
    }

    const int nIdx = lane & 15;
    const int mOff = (lane < 16) ? 0 : 8;
    #pragma unroll
    for (int c = 0; c < 8; ++c) {
        int d = d_base + c * 16 + nIdx;
        float bo = b_out[e * D_DIM + d];
        #pragma unroll
        for (int i = 0; i < 8; ++i) {
            int rloc = wv * 16 + mOff + i;
            int row  = m0 + rloc;
            if (row < cnt) {
                float y = (acc[c][i] + bo) * sWt[rloc];
                slot_out[(size_t)sSlot[rloc] * N_TOK * D_DIM +
                         (size_t)sTok[rloc] * D_DIM + d] = y;
            }
        }
    }
}

__global__ __launch_bounds__(256) void k_combine(const float* __restrict__ s0,
                                                 const float* __restrict__ s1,
                                                 float* __restrict__ out) {
    size_t i = ((size_t)blockIdx.x * blockDim.x + threadIdx.x) * 4;
    float4 a = *(const float4*)(s0 + i);
    float4 b = *(const float4*)(s1 + i);
    float4 o;
    o.x = a.x + b.x; o.y = a.y + b.y; o.z = a.z + b.z; o.w = a.w + b.w;
    *(float4*)(out + i) = o;
}

// ---------------- launch ----------------
extern "C" void kernel_launch(void* const* d_in, const int* in_sizes, int n_in,
                              void* d_out, int out_size, void* d_ws, size_t ws_size,
                              hipStream_t stream) {
    const float* states  = (const float*)d_in[0];
    const float* Wg      = (const float*)d_in[1];
    const float* W_in    = (const float*)d_in[2];
    const float* b_in    = (const float*)d_in[3];
    const float* W_scale = (const float*)d_in[4];
    const float* b_scale = (const float*)d_in[5];
    const float* W_out   = (const float*)d_in[6];
    const float* b_out   = (const float*)d_in[7];
    (void)in_sizes; (void)n_in; (void)out_size; (void)ws_size;

    char* ws = (char*)d_ws;
    int*    counts   = (int*)(ws + OFF_COUNTS);
    int*    offs     = (int*)(ws + OFF_OFFSETS);
    int*    tok      = (int*)(ws + OFF_TOK);
    int*    slot     = (int*)(ws + OFF_SLOT);
    float*  wgt      = (float*)(ws + OFF_WGT);
    __bf16* xbf      = (__bf16*)(ws + OFF_XBF);
    __bf16* inner    = (__bf16*)(ws + OFF_INNER);
    float*  slot_out = (float*)(ws + OFF_SLOTOUT);
    __bf16* WiT      = (__bf16*)(ws + OFF_WIT);
    __bf16* WsT      = (__bf16*)(ws + OFF_WST);
    __bf16* WoT      = (__bf16*)(ws + OFF_WOT);
    float*  out      = (float*)d_out;

    k_init<<<1, 32, 0, stream>>>(counts);
    k_convert<<<(N_TOK * D_DIM / 4) / 256, 256, 0, stream>>>(states, xbf);
    // W_in/W_scale: [D][F] f32 -> [F][D] bf16 ; W_out: [F][D] f32 -> [D][F] bf16
    k_transpose_cvt<<<dim3(F_DIM / 32, D_DIM / 32, E_NUM), 256, 0, stream>>>(W_in, WiT, D_DIM, F_DIM);
    k_transpose_cvt<<<dim3(F_DIM / 32, D_DIM / 32, E_NUM), 256, 0, stream>>>(W_scale, WsT, D_DIM, F_DIM);
    k_transpose_cvt<<<dim3(D_DIM / 32, F_DIM / 32, E_NUM), 256, 0, stream>>>(W_out, WoT, F_DIM, D_DIM);
    k_router<<<N_TOK / 8, 256, 0, stream>>>(states, Wg, counts, tok, slot, wgt);
    k_scan<<<1, 32, 0, stream>>>(counts, offs);
    k_passA<<<dim3(F_DIM / 128, N_TOK / 128, E_NUM), 256, 0, stream>>>(
        WiT, b_in, WsT, b_scale, xbf, counts, offs, tok, inner);
    k_passB<<<dim3(D_DIM / 128, N_TOK / 128, E_NUM), 256, 0, stream>>>(
        WoT, b_out, inner, counts, offs, tok, slot, wgt, slot_out);
    k_combine<<<(N_TOK * D_DIM / 4) / 256, 256, 0, stream>>>(
        slot_out, slot_out + (size_t)N_TOK * D_DIM, out);
}
